// GATBasedMolecularGraphNeuralNetwork_54872502173926
// MI455X (gfx1250) — matrix operand order repaired and checked by reading, not compiled, not run on hardware
//
#include <hip/hip_runtime.h>
#include <hip/hip_bf16.h>
#include <stdint.h>

#define N_NODES 262144
#define N_EDGES 1048576
#define N_GRAPHS 5000
#define IN_DIM 48
#define HID 128

typedef __attribute__((ext_vector_type(16))) _Float16 v16h;
typedef __attribute__((ext_vector_type(8)))  _Float16 v8h;
typedef __attribute__((ext_vector_type(4)))  _Float16 v4h;
typedef __attribute__((ext_vector_type(8)))  float    v8f;
typedef unsigned int u32x4 __attribute__((ext_vector_type(4)));
typedef int          i32x8 __attribute__((ext_vector_type(8)));
typedef int          i32x4 __attribute__((ext_vector_type(4)));

// LDS row stride in halves: 40 halves = 80 bytes (16B aligned, 20-bank lane
// stride -> conflict-free b128 fragment reads across a 16-lane group)
#define LDSTRIDE 40

// Tensor Data Mover: ROCm 7.2 (clang-22) ships a 5-arg builtin, the upstream
// amdgpu-toolchain (clang-23) a 6-arg one (probe-verified arities/types).
#if defined(__HIP_DEVICE_COMPILE__) && \
    __has_builtin(__builtin_amdgcn_tensor_load_to_lds)
#if (__clang_major__ <= 22)
#define TDM_ARITY 5
#else
#define TDM_ARITY 6
#endif
#else
#define TDM_ARITY 0
#endif

// ---------- order-preserving float <-> uint mapping for atomicMax ----------
__device__ __forceinline__ unsigned fenc(float f) {
    unsigned u = __float_as_uint(f);
    return (u & 0x80000000u) ? ~u : (u | 0x80000000u);
}
__device__ __forceinline__ float fdec(unsigned u) {
    unsigned b = (u & 0x80000000u) ? (u & 0x7FFFFFFFu) : ~u;
    return __uint_as_float(b);
}

__device__ __forceinline__ float mishf(float x) {
    float sp = (x > 20.f) ? x : log1pf(expf(x));
    return x * tanhf(sp);
}

// ---------------------------------------------------------------------------
// vectorized fill (graph-capture-safe memset)
// ---------------------------------------------------------------------------
__global__ void fill_u32x4(uint4* __restrict__ p, unsigned v, size_t n4) {
    size_t i = (size_t)blockIdx.x * blockDim.x + threadIdx.x;
    if (i < n4) p[i] = make_uint4(v, v, v, v);
}

// f32 -> f16 conversion (vectorized), n4 = n/4
__global__ void conv_f16(const float* __restrict__ in, _Float16* __restrict__ out,
                         size_t n4) {
    size_t i = (size_t)blockIdx.x * blockDim.x + threadIdx.x;
    if (i >= n4) return;
    float4 v = *(const float4*)&in[i * 4];
    v4h h;
    h[0] = (_Float16)v.x; h[1] = (_Float16)v.y;
    h[2] = (_Float16)v.z; h[3] = (_Float16)v.w;
    *(v4h*)&out[i * 4] = h;
}

// ---------------------------------------------------------------------------
// WMMA GEMM: Y[N, HID] = X[N, C] * W[C, HID]   (C = 48 or 128, HID = 128)
// X, W pre-converted to f16. Block = 256 threads (8 waves); block tile
// 128 rows x 128 cols; wave w owns rows [16w,16w+16) x all 8 col tiles.
// A chunk (128 x 32 halves) staged to LDS by the Tensor Data Mover when
// available (pad fields reproduce the 80B padded row stride; tensor_dim0
// gives OOB zero-fill for the C=48 tail); otherwise manual v8h copies.
// Fragment layouts per CDNA5 ISA 7.12.2 (wave32):
//   A 16x32 f16 : lane l holds row M=l%16; elem i -> K = (i<8?i:i+8) + (l>=16?8:0)
//   B 32x16 f16 : lane l holds col N=l%16; elem i -> K = (i<8?i:i+8) + (l>=16?8:0)
//   D 16x16 f32 : lane l col N=l%16; vgpr j -> row M = j + (l>=16?8:0)
// ---------------------------------------------------------------------------
template <int C>
__global__ void __launch_bounds__(256) gemm_wmma(const _Float16* __restrict__ X,
                                                 const _Float16* __restrict__ W,
                                                 float* __restrict__ Y) {
    __shared__ _Float16 As[128 * LDSTRIDE];   // [row][k]  (k-chunk local)
    __shared__ _Float16 BsT[128 * LDSTRIDE];  // [col][k]  (transposed)

    const int tid  = threadIdx.x;
    const int wave = tid >> 5;
    const int lane = tid & 31;
    const int row0 = blockIdx.x * 128;

    const v8f vzero = {0.f, 0.f, 0.f, 0.f, 0.f, 0.f, 0.f, 0.f};
    v8f acc[8];
#pragma unroll
    for (int t = 0; t < 8; ++t) acc[t] = vzero;

    constexpr int KSTEPS = (C + 31) / 32;
#pragma unroll
    for (int ks = 0; ks < KSTEPS; ++ks) {
        const int k0 = ks * 32;

#if TDM_ARITY > 0
        // ---- A chunk via Tensor Data Mover (one descriptor per block) ----
        if (wave == 0) {
            const unsigned ldsoff = (unsigned)(size_t)(void*)&As[0];
            const unsigned long long ga =
                (unsigned long long)(const void*)&X[(size_t)row0 * C + k0];
            const unsigned td0 = (unsigned)(C - k0);  // OOB zero-fill past C
            u32x4 g0 = {1u,                       // count=1, user mode
                        ldsoff,                   // lds_addr
                        (unsigned)ga,             // global_addr[31:0]
                        ((unsigned)((ga >> 32) & 0x01FFFFFFull)) | (2u << 30)};
            // group1: data_size=2B, pad_enable, pad every 16 DW (64B) by 4 DW
            // (16B) -> 80B LDS rows; tile 32 x 128; tensor dims (C-k0) x 128
            i32x8 g1 = {(int)((1u << 16) | (1u << 20) | (3u << 22) | (3u << 25)),
                        (int)((td0 & 0xFFFFu) << 16),
                        (int)(((td0 >> 16) & 0xFFFFu) | (128u << 16)),
                        (int)(32u << 16),
                        (int)128u,                // tile_dim1=128
                        (int)(unsigned)C,         // tensor_dim0_stride lo
                        0, 0};
            i32x4 gz = {0, 0, 0, 0};
#if TDM_ARITY == 5
            __builtin_amdgcn_tensor_load_to_lds(g0, g1, gz, gz, 0);
#else
            i32x8 gz8 = {0, 0, 0, 0, 0, 0, 0, 0};
            __builtin_amdgcn_tensor_load_to_lds(g0, g1, gz, gz, gz8, 0);
#endif
        }
#else
        // ---- manual A staging: 512 x v8h copies across 256 threads ----
#pragma unroll
        for (int j = 0; j < 2; ++j) {
            const int idx = tid + j * 256;   // v8h unit (0..511)
            const int r   = idx >> 2;        // 4 units per 32-half row
            const int u   = idx & 3;
            v8h hv;
#pragma unroll
            for (int q = 0; q < 8; ++q) hv[q] = (_Float16)0.f;
            if (k0 + u * 8 < C)
                hv = *(const v8h*)&X[(size_t)(row0 + r) * C + k0 + u * 8];
            *(v8h*)&As[r * LDSTRIDE + u * 8] = hv;
        }
#endif
        // ---- stage B chunk transposed: 32 K x 128 cols (f16 source) ----
#pragma unroll
        for (int j = 0; j < 2; ++j) {
            const int idx = tid + j * 256;   // v8h unit (0..511)
            const int k   = idx >> 4;        // 16 units per 128-col row
            const int c0  = (idx & 15) * 8;
            v8h bv;
#pragma unroll
            for (int q = 0; q < 8; ++q) bv[q] = (_Float16)0.f;
            if (k0 + k < C)
                bv = *(const v8h*)&W[(size_t)(k0 + k) * HID + c0];
#pragma unroll
            for (int q = 0; q < 8; ++q) BsT[(c0 + q) * LDSTRIDE + k] = bv[q];
        }
#if TDM_ARITY > 0
        __builtin_amdgcn_s_wait_tensorcnt(0);
#endif
        __syncthreads();

        // ---- A fragment: two b128 LDS loads ----
        v16h a;
        {
            const int r    = wave * 16 + (lane & 15);
            const int koff = (lane >= 16) ? 8 : 0;
            v8h alo = *(const v8h*)&As[r * LDSTRIDE + koff];
            v8h ahi = *(const v8h*)&As[r * LDSTRIDE + 16 + koff];
#pragma unroll
            for (int i = 0; i < 8; ++i) { a[i] = alo[i]; a[i + 8] = ahi[i]; }
        }
        const int koffB = (lane >= 16) ? 8 : 0;
#pragma unroll
        for (int ct = 0; ct < 8; ++ct) {
            const int col = ct * 16 + (lane & 15);
            v8h blo = *(const v8h*)&BsT[col * LDSTRIDE + koffB];
            v8h bhi = *(const v8h*)&BsT[col * LDSTRIDE + koffB + 16];
            v16h b;
#pragma unroll
            for (int i = 0; i < 8; ++i) { b[i] = blo[i]; b[i + 8] = bhi[i]; }
            acc[ct] = __builtin_amdgcn_wmma_f32_16x16x32_f16(
                false, a, false, b, (short)0, acc[ct], false, false);
        }
        __syncthreads();
    }

    // ---- store D tiles (coalesced across lanes) ----
    const int n     = lane & 15;
    const int mbase = (lane >= 16) ? 8 : 0;
#pragma unroll
    for (int ct = 0; ct < 8; ++ct) {
#pragma unroll
        for (int j = 0; j < 8; ++j) {
            Y[(size_t)(row0 + wave * 16 + mbase + j) * HID + ct * 16 + n] =
                acc[ct][j];
        }
    }
}

// ---------------------------------------------------------------------------
// Edge stage: one wave per edge, 4 channels per lane, shuffle reduction.
// Segment-max fused in (lane 0 atomicMax on the order-preserving encoding).
// ---------------------------------------------------------------------------
__global__ void __launch_bounds__(256) edge_logits(
    const float* __restrict__ xl, const float* __restrict__ xr,
    const float* __restrict__ ea, const float* __restrict__ We,
    const float* __restrict__ att,
    const int* __restrict__ src, const int* __restrict__ dst,
    float* __restrict__ logits, unsigned* __restrict__ m_u) {
    const int wave = threadIdx.x >> 5;
    const int lane = threadIdx.x & 31;
    const int e    = blockIdx.x * 8 + wave;
    const int s = src[e], d = dst[e];
    const int c0 = lane * 4;

    float4 vl = *(const float4*)&xl[(size_t)s * HID + c0];
    float4 vr = *(const float4*)&xr[(size_t)d * HID + c0];
    float eav[6];
#pragma unroll
    for (int k = 0; k < 6; ++k) eav[k] = ea[(size_t)e * 6 + k];

    float acc = 0.f;
#pragma unroll
    for (int j = 0; j < 4; ++j) {
        const int c = c0 + j;
        float w = 0.f;
#pragma unroll
        for (int k = 0; k < 6; ++k) w += eav[k] * We[k * HID + c];
        float v = (&vl.x)[j] + (&vr.x)[j] + w;
        v = (v > 0.f) ? v : 0.2f * v;          // leaky_relu(0.2)
        acc += v * att[c];
    }
#pragma unroll
    for (int off = 16; off > 0; off >>= 1) acc += __shfl_xor(acc, off, 32);
    if (lane == 0) {
        logits[e] = acc;
        atomicMax(&m_u[d], fenc(acc));
    }
}

__global__ void edge_sumexp(const float* __restrict__ logits,
                            const int* __restrict__ dst,
                            const unsigned* __restrict__ m_u,
                            float* __restrict__ ssum) {
    const int e = blockIdx.x * blockDim.x + threadIdx.x;
    const int d = dst[e];
    atomicAdd(&ssum[d], expf(logits[e] - fdec(m_u[d])));
}

__global__ void __launch_bounds__(256) edge_scatter(
    const float* __restrict__ logits, const float* __restrict__ xl,
    const int* __restrict__ src, const int* __restrict__ dst,
    const unsigned* __restrict__ m_u, const float* __restrict__ ssum,
    float* __restrict__ out) {
    const int wave = threadIdx.x >> 5;
    const int lane = threadIdx.x & 31;
    const int e    = blockIdx.x * 8 + wave;
    const int s = src[e], d = dst[e];
    const int c0 = lane * 4;
    const float alpha = expf(logits[e] - fdec(m_u[d])) / (ssum[d] + 1e-16f);
    float4 v = *(const float4*)&xl[(size_t)s * HID + c0];
    float* o = &out[(size_t)d * HID + c0];
    atomicAdd(o + 0, alpha * v.x);
    atomicAdd(o + 1, alpha * v.y);
    atomicAdd(o + 2, alpha * v.z);
    atomicAdd(o + 3, alpha * v.w);
}

// ---------------------------------------------------------------------------
// BatchNorm stats / apply + PReLU  (bias cancels under mean-subtract).
// Apply kernels also emit the f16 copy of h for the next layer's WMMA GEMM.
// ---------------------------------------------------------------------------
__global__ void __launch_bounds__(256) bn_stats(const float* __restrict__ A,
                                                float* __restrict__ stats) {
    const int c   = threadIdx.x & 127;
    const int sub = threadIdx.x >> 7;   // 0 or 1
    float s = 0.f, q = 0.f;
    for (int row = blockIdx.x * 2 + sub; row < N_NODES; row += gridDim.x * 2) {
        float v = A[(size_t)row * HID + c];
        s += v;
        q += v * v;
    }
    __shared__ float red[256];
    red[threadIdx.x] = s;
    __syncthreads();
    if (sub == 0) atomicAdd(&stats[c], red[c] + red[128 + c]);
    __syncthreads();
    red[threadIdx.x] = q;
    __syncthreads();
    if (sub == 0) atomicAdd(&stats[128 + c], red[c] + red[128 + c]);
}

__global__ void bn_apply(const float* __restrict__ A,
                         const float* __restrict__ stats,
                         const float* __restrict__ gamma,
                         const float* __restrict__ beta,
                         const float* __restrict__ prelu,
                         float* __restrict__ H, _Float16* __restrict__ H16) {
    const size_t i4 = (size_t)blockIdx.x * blockDim.x + threadIdx.x;
    const int c0 = (int)(i4 & 31) * 4;
    const float inv_n = 1.0f / (float)N_NODES;
    const float p = prelu[0];
    float4 v = *(const float4*)&A[i4 * 4];
    float4 r;
    v4h h;
#pragma unroll
    for (int j = 0; j < 4; ++j) {
        const int c = c0 + j;
        const float mu  = stats[c] * inv_n;
        const float var = stats[128 + c] * inv_n - mu * mu;
        float y = gamma[c] * ((&v.x)[j] - mu) * rsqrtf(var + 1e-5f) + beta[c];
        y = (y > 0.f) ? y : p * y;
        (&r.x)[j] = y;
        h[j] = (_Float16)y;
    }
    *(float4*)&H[i4 * 4] = r;
    *(v4h*)&H16[i4 * 4] = h;
}

__global__ void bias_add(const float* __restrict__ A,
                         const float* __restrict__ bias,
                         float* __restrict__ H) {
    const size_t i4 = (size_t)blockIdx.x * blockDim.x + threadIdx.x;
    const int c0 = (int)(i4 & 31) * 4;
    float4 v = *(const float4*)&A[i4 * 4];
    float4 b = *(const float4*)&bias[c0];
    v.x += b.x; v.y += b.y; v.z += b.z; v.w += b.w;
    *(float4*)&H[i4 * 4] = v;
}

// ---------------------------------------------------------------------------
// Pooling: per-(graph,channel) softmax aggregation + global add pool
// ---------------------------------------------------------------------------
__global__ void pool_max(const float* __restrict__ H, const int* __restrict__ batch,
                         const float* __restrict__ tptr, unsigned* __restrict__ wmax) {
    const size_t i4 = (size_t)blockIdx.x * blockDim.x + threadIdx.x;
    const int node = (int)(i4 >> 5), c0 = (int)(i4 & 31) * 4;
    const int b = batch[node];
    const float t = tptr[0];
    float4 h = *(const float4*)&H[i4 * 4];
    unsigned* m = &wmax[(size_t)b * 128 + c0];
#pragma unroll
    for (int j = 0; j < 4; ++j) atomicMax(m + j, fenc((&h.x)[j] * t));
}

__global__ void pool_sum(const float* __restrict__ H, const int* __restrict__ batch,
                         const float* __restrict__ tptr, const unsigned* __restrict__ wmax,
                         float* __restrict__ wsum, float* __restrict__ p2) {
    const size_t i4 = (size_t)blockIdx.x * blockDim.x + threadIdx.x;
    const int node = (int)(i4 >> 5), c0 = (int)(i4 & 31) * 4;
    const int b = batch[node];
    const float t = tptr[0];
    float4 h = *(const float4*)&H[i4 * 4];
    const size_t gb = (size_t)b * 128 + c0;
#pragma unroll
    for (int j = 0; j < 4; ++j) {
        const float hv = (&h.x)[j];
        atomicAdd(&wsum[gb + j], expf(hv * t - fdec(wmax[gb + j])));
        atomicAdd(&p2[gb + j], hv);
    }
}

__global__ void pool_weighted(const float* __restrict__ H, const int* __restrict__ batch,
                              const float* __restrict__ tptr, const unsigned* __restrict__ wmax,
                              const float* __restrict__ wsum, float* __restrict__ p1) {
    const size_t i4 = (size_t)blockIdx.x * blockDim.x + threadIdx.x;
    const int node = (int)(i4 >> 5), c0 = (int)(i4 & 31) * 4;
    const int b = batch[node];
    const float t = tptr[0];
    float4 h = *(const float4*)&H[i4 * 4];
    const size_t gb = (size_t)b * 128 + c0;
#pragma unroll
    for (int j = 0; j < 4; ++j) {
        const float hv = (&h.x)[j];
        const float w = expf(hv * t - fdec(wmax[gb + j])) / (wsum[gb + j] + 1e-16f);
        atomicAdd(&p1[gb + j], w * hv);
    }
}

// ---------------------------------------------------------------------------
// Final MLP: z[256] -> mish(fc 128) -> mish(fc2 64) -> A (3)
// ---------------------------------------------------------------------------
__global__ void __launch_bounds__(128) mlp(
    const float* __restrict__ p1, const float* __restrict__ p2,
    const float* __restrict__ fcW, const float* __restrict__ fcb,
    const float* __restrict__ fc2W, const float* __restrict__ fc2b,
    const float* __restrict__ AW, const float* __restrict__ Ab,
    float* __restrict__ out) {
    const int g = blockIdx.x, t = threadIdx.x;
    __shared__ float z[256];
    __shared__ float h1[128];
    __shared__ float h2[64];
    z[t]       = p1[(size_t)g * 128 + t];
    z[128 + t] = p2[(size_t)g * 128 + t];
    __syncthreads();
    float a = fcb[t];
    for (int k = 0; k < 256; ++k) a += z[k] * fcW[k * 128 + t];
    h1[t] = mishf(a);
    __syncthreads();
    if (t < 64) {
        float a2 = fc2b[t];
        for (int k = 0; k < 128; ++k) a2 += h1[k] * fc2W[k * 64 + t];
        h2[t] = mishf(a2);
    }
    __syncthreads();
    if (t < 3) {
        float a3 = Ab[t];
        for (int k = 0; k < 64; ++k) a3 += h2[k] * AW[k * 3 + t];
        out[(size_t)g * 3 + t] = a3;
    }
}

// ---------------------------------------------------------------------------
extern "C" void kernel_launch(void* const* d_in, const int* in_sizes, int n_in,
                              void* d_out, int out_size, void* d_ws, size_t ws_size,
                              hipStream_t stream) {
    (void)in_sizes; (void)n_in; (void)out_size; (void)ws_size;

    const float* x     = (const float*)d_in[0];
    const int*   ei    = (const int*)d_in[1];
    const int*   src   = ei;
    const int*   dst   = ei + N_EDGES;
    const float* ea    = (const float*)d_in[2];
    const int*   batch = (const int*)d_in[3];
    // params: 6 layers x {Wl, Wr, We, att, bias, gamma, beta, prelu} = idx 4..51
    const float* softmax_t = (const float*)d_in[52];
    const float* fcW  = (const float*)d_in[53];
    const float* fcb  = (const float*)d_in[54];
    const float* fc2W = (const float*)d_in[55];
    const float* fc2b = (const float*)d_in[56];
    const float* AW   = (const float*)d_in[57];
    const float* Ab   = (const float*)d_in[58];

    // ---- workspace carve-out ----
    char* ws = (char*)d_ws;
    size_t off = 0;
    auto take = [&](size_t bytes) -> void* {
        void* p = ws + off;
        off += (bytes + 255) & ~(size_t)255;
        return p;
    };
    float*     hbuf   = (float*)take((size_t)N_NODES * HID * 4);
    float*     xl     = (float*)take((size_t)N_NODES * HID * 4);
    float*     xr     = (float*)take((size_t)N_NODES * HID * 4);
    float*     outacc = (float*)take((size_t)N_NODES * HID * 4);
    _Float16*  hh16   = (_Float16*)take((size_t)N_NODES * HID * 2);
    _Float16*  x16    = (_Float16*)take((size_t)N_NODES * IN_DIM * 2);
    _Float16*  wl16   = (_Float16*)take((size_t)HID * HID * 2);
    _Float16*  wr16   = (_Float16*)take((size_t)HID * HID * 2);
    float*     logits = (float*)take((size_t)N_EDGES * 4);
    unsigned*  m_u    = (unsigned*)take((size_t)N_NODES * 4);
    float*     ssum   = (float*)take((size_t)N_NODES * 4);
    float*     stats  = (float*)take(256 * 4);
    unsigned*  wmax   = (unsigned*)take((size_t)N_GRAPHS * 128 * 4);
    float*     wsum   = (float*)take((size_t)N_GRAPHS * 128 * 4);
    float*     p1     = (float*)take((size_t)N_GRAPHS * 128 * 4);
    float*     p2     = (float*)take((size_t)N_GRAPHS * 128 * 4);

    const size_t NH = (size_t)N_NODES * HID;   // 33,554,432
    const size_t GH = (size_t)N_GRAPHS * 128;  // 640,000

    // x -> f16 once (layer-0 GEMM input)
    conv_f16<<<(unsigned)((size_t)N_NODES * IN_DIM / 4 / 256), 256, 0, stream>>>(
        x, x16, (size_t)N_NODES * IN_DIM / 4);

    for (int l = 0; l < 6; ++l) {
        const int base = 4 + l * 8;
        const float* Wl    = (const float*)d_in[base + 0];
        const float* Wr    = (const float*)d_in[base + 1];
        const float* We    = (const float*)d_in[base + 2];
        const float* att   = (const float*)d_in[base + 3];
        const float* bias  = (const float*)d_in[base + 4];
        const float* gamma = (const float*)d_in[base + 5];
        const float* beta  = (const float*)d_in[base + 6];
        const float* prelu = (const float*)d_in[base + 7];

        const int C = (l == 0) ? IN_DIM : HID;
        const unsigned wblocks = (unsigned)((size_t)C * HID / 4 / 256);
        conv_f16<<<wblocks ? wblocks : 1, 256, 0, stream>>>(Wl, wl16, (size_t)C * HID / 4);
        conv_f16<<<wblocks ? wblocks : 1, 256, 0, stream>>>(Wr, wr16, (size_t)C * HID / 4);

        if (l == 0) {
            gemm_wmma<IN_DIM><<<N_NODES / 128, 256, 0, stream>>>(x16, wl16, xl);
            gemm_wmma<IN_DIM><<<N_NODES / 128, 256, 0, stream>>>(x16, wr16, xr);
        } else {
            gemm_wmma<HID><<<N_NODES / 128, 256, 0, stream>>>(hh16, wl16, xl);
            gemm_wmma<HID><<<N_NODES / 128, 256, 0, stream>>>(hh16, wr16, xr);
        }

        fill_u32x4<<<N_NODES / 4 / 256, 256, 0, stream>>>((uint4*)m_u, 0u, (size_t)N_NODES / 4);
        fill_u32x4<<<N_NODES / 4 / 256, 256, 0, stream>>>((uint4*)ssum, 0u, (size_t)N_NODES / 4);
        fill_u32x4<<<(unsigned)(NH / 4 / 256), 256, 0, stream>>>((uint4*)outacc, 0u, NH / 4);

        edge_logits<<<N_EDGES / 8, 256, 0, stream>>>(xl, xr, ea, We, att, src, dst,
                                                     logits, m_u);
        edge_sumexp<<<N_EDGES / 256, 256, 0, stream>>>(logits, dst, m_u, ssum);
        edge_scatter<<<N_EDGES / 8, 256, 0, stream>>>(logits, xl, src, dst, m_u, ssum, outacc);

        if (l < 5) {
            fill_u32x4<<<1, 64, 0, stream>>>((uint4*)stats, 0u, (size_t)64);
            bn_stats<<<512, 256, 0, stream>>>(outacc, stats);
            bn_apply<<<(unsigned)(NH / 4 / 256), 256, 0, stream>>>(outacc, stats, gamma,
                                                                   beta, prelu, hbuf, hh16);
        } else {
            bias_add<<<(unsigned)(NH / 4 / 256), 256, 0, stream>>>(outacc, bias, hbuf);
        }
    }

    // ---- pooling ----
    fill_u32x4<<<(unsigned)(GH / 4 / 256), 256, 0, stream>>>((uint4*)wmax, 0u, GH / 4);
    fill_u32x4<<<(unsigned)(GH / 4 / 256), 256, 0, stream>>>((uint4*)wsum, 0u, GH / 4);
    fill_u32x4<<<(unsigned)(GH / 4 / 256), 256, 0, stream>>>((uint4*)p1, 0u, GH / 4);
    fill_u32x4<<<(unsigned)(GH / 4 / 256), 256, 0, stream>>>((uint4*)p2, 0u, GH / 4);

    pool_max<<<(unsigned)(NH / 4 / 256), 256, 0, stream>>>(hbuf, batch, softmax_t, wmax);
    pool_sum<<<(unsigned)(NH / 4 / 256), 256, 0, stream>>>(hbuf, batch, softmax_t, wmax, wsum, p2);
    pool_weighted<<<(unsigned)(NH / 4 / 256), 256, 0, stream>>>(hbuf, batch, softmax_t, wmax, wsum, p1);

    mlp<<<N_GRAPHS, 128, 0, stream>>>(p1, p2, fcW, fcb, fc2W, fc2b, AW, Ab, (float*)d_out);
}
